// SingleAttention_68650757260115
// MI455X (gfx1250) — matrix operand order repaired and checked by reading, not compiled, not run on hardware
//
#include <hip/hip_runtime.h>

typedef __attribute__((ext_vector_type(16))) _Float16 v16h;
typedef __attribute__((ext_vector_type(8)))  _Float16 v8h;
typedef __attribute__((ext_vector_type(2)))  _Float16 h2;
typedef __attribute__((ext_vector_type(8)))  float    v8f;

#define WMMA_F16(A, B, C) \
  __builtin_amdgcn_wmma_f32_16x16x32_f16(false, (A), false, (B), (short)0, (C), false, false)

// DPP lane-permute within a 16-lane row (wave32): cheap VALU, no LDS traffic.
#define DPP_XOR1  0xB1   // quad_perm(1,0,3,2)
#define DPP_XOR2  0x4E   // quad_perm(2,3,0,1)
#define DPP_HMIRR 0x141  // row_half_mirror (lane ^ 7 within 8)
#define DPP_MIRR  0x140  // row_mirror      (lane ^ 15 within 16)

template <int CTRL>
__device__ __forceinline__ float dpp_movf(float x) {
  int r = __builtin_amdgcn_update_dpp(0, __builtin_bit_cast(int, x),
                                      CTRL, 0xf, 0xf, true);
  return __builtin_bit_cast(float, r);
}

__device__ __forceinline__ float row_reduce_max(float t) {
  t = fmaxf(t, dpp_movf<DPP_XOR1>(t));
  t = fmaxf(t, dpp_movf<DPP_XOR2>(t));
  t = fmaxf(t, dpp_movf<DPP_HMIRR>(t));
  t = fmaxf(t, dpp_movf<DPP_MIRR>(t));
  return t;
}

__device__ __forceinline__ float row_reduce_sum(float t) {
  t += dpp_movf<DPP_XOR1>(t);
  t += dpp_movf<DPP_XOR2>(t);
  t += dpp_movf<DPP_HMIRR>(t);
  t += dpp_movf<DPP_MIRR>(t);
  return t;
}

static constexpr int S_LEN = 2048;
static constexpr int D_DIM = 64;
static constexpr int QB    = 64;   // query rows per workgroup (16 per wave, 4 waves)
static constexpr int KB    = 64;   // key block: one softmax update per 64 keys
static constexpr int KPAD  = 72;   // halves per K-tile row in LDS (64 + 8 pad)
static constexpr int VPAD  = 72;   // halves per Vt row in LDS (64 keys + 8 pad)
static constexpr int PPAD  = 72;   // halves per P row in LDS (64 keys + 8 pad)

__global__ __launch_bounds__(128)
void fa_causal_kernel(const float* __restrict__ q, const float* __restrict__ k,
                      const float* __restrict__ v, float* __restrict__ out) {
  __shared__ _Float16 Klds[KB * KPAD];      // K tile, row-major [key][d]
  __shared__ _Float16 Vtlds[D_DIM * VPAD];  // V tile, transposed [d][key]
  __shared__ _Float16 Plds[4][16 * PPAD];   // per-wave P scratch [q][key]

  const int tid  = threadIdx.x;
  const int w    = tid >> 5;
  const int lane = tid & 31;
  const int ln   = lane & 15;
  const bool hi  = lane >= 16;
  const int qb   = blockIdx.x * QB;
  const size_t base = (size_t)blockIdx.y * S_LEN * D_DIM;

  // ---- Q A-fragments, pre-scaled by 1/sqrt(D) * log2(e) (softmax in base 2) ----
  const float qscale = 0.125f * 1.44269504088896f;
  const int k0 = hi ? 8 : 0;                     // A layout: K-base per lane half
  v16h a_lo, a_hi;
  {
    const float* qp = q + base + (size_t)(qb + w * 16 + ln) * D_DIM + k0;
#pragma unroll
    for (int e = 0; e < 8; ++e) {
      a_lo[e]     = (_Float16)(qp[e]      * qscale);  // K = k0 + e
      a_lo[e + 8] = (_Float16)(qp[e + 16] * qscale);  // K = k0 + 16 + e
      a_hi[e]     = (_Float16)(qp[e + 32] * qscale);  // K = 32 + k0 + e
      a_hi[e + 8] = (_Float16)(qp[e + 48] * qscale);  // K = 32 + k0 + 16 + e
    }
  }

  v8f o[4] = {v8f{}, v8f{}, v8f{}, v8f{}};       // O accum: d-chunks 0..3
  float m[8], l[8];
#pragma unroll
  for (int j = 0; j < 8; ++j) { m[j] = -1e30f; l[j] = 0.0f; }

  const int kbend = qb + QB;                     // causal: keys up to last q row

  // staging decompositions (128 threads, 64x64 tiles)
  const int krow  = tid >> 1;                    // K: key row 0..63
  const int kdseg = (tid & 1) << 5;              // K: d segment 0 / 32
  const int vkp   = (tid & 31) << 1;             // V: even key pair 0..62
  const int vdseg = (tid >> 5) << 4;             // V: d segment 0,16,32,48

  for (int kb = 0; kb < kbend; kb += KB) {
    __syncthreads();  // previous iteration's readers done before restaging
    // ---- cooperative staging ----
    {
      // K tile: row-major f16, contiguous stores (packed cvt + wide ds stores)
      const float* kp = k + base + (size_t)(kb + krow) * D_DIM + kdseg;
      float kv[32];
#pragma unroll
      for (int c = 0; c < 8; ++c) *(float4*)&kv[c * 4] = ((const float4*)kp)[c];
      _Float16* kd = &Klds[krow * KPAD + kdseg];
#pragma unroll
      for (int i = 0; i < 32; ++i) kd[i] = (_Float16)kv[i];

      // V tile: transposed f16; key-pair per thread -> half2 stores along keys
      const float* vp0 = v + base + (size_t)(kb + vkp) * D_DIM + vdseg;
      const float* vp1 = vp0 + D_DIM;
      float va[16], vb[16];
#pragma unroll
      for (int c = 0; c < 4; ++c) {
        *(float4*)&va[c * 4] = ((const float4*)vp0)[c];
        *(float4*)&vb[c * 4] = ((const float4*)vp1)[c];
      }
#pragma unroll
      for (int i = 0; i < 16; ++i) {
        h2 pairv = {(_Float16)va[i], (_Float16)vb[i]};
        *(h2*)&Vtlds[(vdseg + i) * VPAD + vkp] = pairv;
      }

      if (kb + KB < kbend) {                     // L2 prefetch of next tiles
        __builtin_prefetch(kp + (size_t)KB * D_DIM, 0, 1);
        __builtin_prefetch(vp0 + (size_t)KB * D_DIM, 0, 1);
      }
    }
    __syncthreads();

    // ---- scores: S = Q K^T for four 16-key tiles ----
    v8f s[4];
#pragma unroll
    for (int h4 = 0; h4 < 4; ++h4) {
      const int key = h4 * 16 + ln;              // B layout: N = key = lane&15
      const int db  = hi ? 8 : 0;                // K(=d) base per lane half
      const _Float16* kr = &Klds[key * KPAD + db];
      v8h t0 = *(const v8h*)(kr);
      v8h t1 = *(const v8h*)(kr + 16);
      v8h t2 = *(const v8h*)(kr + 32);
      v8h t3 = *(const v8h*)(kr + 48);
      v16h b_lo = __builtin_shufflevector(t0, t1, 0, 1, 2, 3, 4, 5, 6, 7,
                                          8, 9, 10, 11, 12, 13, 14, 15);
      v16h b_hi = __builtin_shufflevector(t2, t3, 0, 1, 2, 3, 4, 5, 6, 7,
                                          8, 9, 10, 11, 12, 13, 14, 15);
      v8f c = {};
      c = WMMA_F16(a_lo, b_lo, c);
      c = WMMA_F16(a_hi, b_hi, c);
      s[h4] = c;
    }

    // ---- causal mask (diagonal block only: the last one) ----
    const int qg = qb + w * 16 + (hi ? 8 : 0);   // row M = qg + j
    if (kb + KB - 1 > qb + w * 16) {
#pragma unroll
      for (int h4 = 0; h4 < 4; ++h4) {
#pragma unroll
        for (int j = 0; j < 8; ++j) {
          if (kb + h4 * 16 + ln > qg + j) s[h4][j] = -1e30f;
        }
      }
    }

    // ---- online softmax: one update per 64 keys (DPP row reductions) ----
#pragma unroll
    for (int j = 0; j < 8; ++j) {
      float t = fmaxf(fmaxf(s[0][j], s[1][j]), fmaxf(s[2][j], s[3][j]));
      t = row_reduce_max(t);
      const float mn = fmaxf(m[j], t);
      const float f  = __builtin_amdgcn_exp2f(m[j] - mn);
      s[0][j] = __builtin_amdgcn_exp2f(s[0][j] - mn);
      s[1][j] = __builtin_amdgcn_exp2f(s[1][j] - mn);
      s[2][j] = __builtin_amdgcn_exp2f(s[2][j] - mn);
      s[3][j] = __builtin_amdgcn_exp2f(s[3][j] - mn);
      const float r = row_reduce_sum((s[0][j] + s[1][j]) + (s[2][j] + s[3][j]));
      l[j] = l[j] * f + r;
      m[j] = mn;
      o[0][j] *= f; o[1][j] *= f; o[2][j] *= f; o[3][j] *= f;
    }

    // ---- P: C-layout -> A-layout via per-wave LDS ----
    _Float16* pw = Plds[w];
    const int prow0 = hi ? 8 : 0;
#pragma unroll
    for (int j = 0; j < 8; ++j) {
      pw[(prow0 + j) * PPAD + ln]      = (_Float16)s[0][j];
      pw[(prow0 + j) * PPAD + 16 + ln] = (_Float16)s[1][j];
      pw[(prow0 + j) * PPAD + 32 + ln] = (_Float16)s[2][j];
      pw[(prow0 + j) * PPAD + 48 + ln] = (_Float16)s[3][j];
    }
    asm volatile("s_wait_dscnt 0" ::: "memory");

    v16h pa0, pa1;                               // P A-frags: keys 0..31, 32..63
    {
      const _Float16* pr = &pw[ln * PPAD + k0];
      v8h u0 = *(const v8h*)(pr);
      v8h u1 = *(const v8h*)(pr + 16);
      v8h u2 = *(const v8h*)(pr + 32);
      v8h u3 = *(const v8h*)(pr + 48);
      pa0 = __builtin_shufflevector(u0, u1, 0, 1, 2, 3, 4, 5, 6, 7,
                                    8, 9, 10, 11, 12, 13, 14, 15);
      pa1 = __builtin_shufflevector(u2, u3, 0, 1, 2, 3, 4, 5, 6, 7,
                                    8, 9, 10, 11, 12, 13, 14, 15);
    }

    // ---- O += P V (K contraction = 64 keys, split 2 x 32) ----
    const int kseg = hi ? 8 : 0;
#pragma unroll
    for (int c = 0; c < 4; ++c) {
      const _Float16* vr = &Vtlds[(c * 16 + ln) * VPAD + kseg];
      v8h u0 = *(const v8h*)(vr);
      v8h u1 = *(const v8h*)(vr + 16);
      v8h u2 = *(const v8h*)(vr + 32);
      v8h u3 = *(const v8h*)(vr + 48);
      v16h vb0 = __builtin_shufflevector(u0, u1, 0, 1, 2, 3, 4, 5, 6, 7,
                                         8, 9, 10, 11, 12, 13, 14, 15);
      v16h vb1 = __builtin_shufflevector(u2, u3, 0, 1, 2, 3, 4, 5, 6, 7,
                                         8, 9, 10, 11, 12, 13, 14, 15);
      o[c] = WMMA_F16(pa0, vb0, o[c]);
      o[c] = WMMA_F16(pa1, vb1, o[c]);
    }
  }

  // ---- normalize and store ----
  const int orow0 = qb + w * 16 + (hi ? 8 : 0);
#pragma unroll
  for (int j = 0; j < 8; ++j) {
    const float inv = (l[j] > 0.0f) ? (1.0f / l[j]) : 0.0f;
    float* op = out + base + (size_t)(orow0 + j) * D_DIM + ln;
    op[0]  = o[0][j] * inv;
    op[16] = o[1][j] * inv;
    op[32] = o[2][j] * inv;
    op[48] = o[3][j] * inv;
  }
}

extern "C" void kernel_launch(void* const* d_in, const int* in_sizes, int n_in,
                              void* d_out, int out_size, void* d_ws, size_t ws_size,
                              hipStream_t stream) {
  const float* q = (const float*)d_in[0];
  const float* k = (const float*)d_in[1];
  const float* v = (const float*)d_in[2];
  float* out = (float*)d_out;
  (void)in_sizes; (void)n_in; (void)out_size; (void)d_ws; (void)ws_size;
  dim3 grid(S_LEN / QB, 2 * 16);  // 32 q-blocks x (B*H = 32)
  fa_causal_kernel<<<grid, dim3(128), 0, stream>>>(q, k, v, out);
}
